// VQVAE_12584254177608
// MI455X (gfx1250) — matrix operand order repaired and checked by reading, not compiled, not run on hardware
//
#include <hip/hip_runtime.h>
#include <hip/hip_bf16.h>
#include <math.h>

// ---------------------------------------------------------------------------
// VQ-VAE forward for gfx1250 (MI455X). All matmul-shaped work (3x3/1x1 convs,
// attention projections) runs through an implicit-GEMM kernel built on
// v_wmma_f32_16x16x32_f16 (f16 inputs staged via LDS, f32 accumulate).
// Per iteration each wave issues 4 WMMAs (2 N-subtiles x 2 K-chunks) with
// vectorized (b128) LDS fragment loads; next weight chunk is prefetched via
// global_prefetch_b8. Attention core (dh=8), GroupNorm/SiLU, ConvTranspose
// and quantize are VALU.
//
// Input order assumption (compile-only, unverifiable): d_in[0] = x, then the
// params pytree flattened with sorted dict keys / list order (jax tree order):
//   1..2    dec_gn.{b,g}
//   3..4    dec_in.{b,w}
//   5..8    dec_mid.attn[0].{in_b,in_w,out_b,out_w}
//   9..10   dec_mid.attn_gn[0].{b,g}
//   11..20  dec_mid.res[0].{conv1.b,conv1.w,conv2.b,conv2.w,gn1.b,gn1.g,gn2.b,gn2.g,res.b,res.w}
//   21..30  dec_mid.res[1] (same order)
//   31..32  dec_out.{b,w}
//   33..42  down[0] resblock (same 10-leaf order)
//   43..52  down[1]
//   53      emb (20x3)
//   54..55  enc_gn.{b,g}
//   56..57  enc_in.{b,w}
//   58..61  enc_mid.attn[0].{in_b,in_w,out_b,out_w}
//   62..63  enc_mid.attn_gn[0].{b,g}
//   64..73  enc_mid.res[0]
//   74..83  enc_mid.res[1]
//   84..85  enc_out.{b,w}
//   86..87  post_q.{b,w}
//   88..89  pre_q.{b,w}
//   90..99  up[0].res
//   100..101 up[0].upconv.{b,w}   (w: [in,out,4,4])
//   102..111 up[1].res
//   112..113 up[1].upconv.{b,w}
// d_out: d[2,3,160,160] (153600) | q[2,3,40,40] (9600) | cb_loss | cm_loss
// ---------------------------------------------------------------------------

typedef __attribute__((ext_vector_type(16))) _Float16 v16h;
typedef __attribute__((ext_vector_type(8)))  _Float16 v8h;
typedef __attribute__((ext_vector_type(8)))  float    v8f;

#define NH 16
#define DH 8

// ---------------------------------------------------------------------------
// Implicit-GEMM conv: out[b,co,y,x] = bias[co] + sum_{ci,ky,kx} w*in
// GEMM view: M = Cout, N = B*H*W, K = Cin*k*k.
// Block = 4 waves. Each wave owns a 16(M) x 32(N) output tile; K is consumed
// 64 at a time (2 WMMA chunks) -> 4 v_wmma per stage/barrier round.
// Weight layout [Cout][Cin][k][k] is already row-major over K: w[co*KK + kk].
// ---------------------------------------------------------------------------
__global__ __launch_bounds__(128) void conv_wmma_kernel(
    const float* __restrict__ in, const float* __restrict__ w,
    const float* __restrict__ bias, float* __restrict__ out,
    int B, int Cin, int H, int W, int Cout, int k, int pad)
{
    __shared__ _Float16 Al[16 * 64];        // A: 16 rows (M) x 64 (K)
    __shared__ _Float16 Bl[4][64 * 32];     // B per wave: 64 (K) x 32 (N)

    const int tid  = threadIdx.x;
    const int wid  = tid >> 5;
    const int lane = tid & 31;
    const int HW   = H * W;
    const int Ntot = B * HW;
    const int ksq  = k * k;
    const int KK   = Cin * ksq;
    const int mbase = blockIdx.y * 16;
    const int nwave = (blockIdx.x * 4 + wid) * 32;

    v8f c0 = {}, c1 = {};

    for (int kc = 0; kc < KK; kc += 64) {
        __syncthreads();
        // Prefetch next weight K-chunk into cache (global_prefetch_b8).
        if (tid == 0 && kc + 64 < KK)
            __builtin_prefetch(&w[(size_t)mbase * KK + kc + 64], 0, 1);

        // Stage A (weights): contiguous over K. 1024 halfs / 128 threads.
        for (int e = tid; e < 1024; e += 128) {
            int r = e >> 6, i = e & 63;
            int co = mbase + r, kk = kc + i;
            float v = 0.f;
            if (co < Cout && kk < KK)
                v = w[(size_t)co * KK + kk];
            Al[r * 64 + i] = (_Float16)v;
        }
        // Stage B (im2col) per wave: 2048 halfs / 32 lanes.
        for (int e = lane; e < 2048; e += 32) {
            int i = e >> 5, cc = e & 31;
            int n = nwave + cc, kk = kc + i;
            float v = 0.f;
            if (n < Ntot && kk < KK) {
                int b = n / HW, s = n % HW;
                int y = s / W,  x = s % W;
                int ci = kk / ksq, rem = kk % ksq;
                int ky = rem / k,  kx  = rem % k;
                int iy = y + ky - pad, ix = x + kx - pad;
                if (iy >= 0 && iy < H && ix >= 0 && ix < W)
                    v = in[(((size_t)b * Cin + ci) * H + iy) * W + ix];
            }
            Bl[wid][cc * 64 + i] = (_Float16)v;
        }
        __syncthreads();

        // Fragments per ISA 16-bit wave32 layouts.
        // A 16x32: lane L<16 -> M=L, halves 0..7 = K 0..7, 8..15 = K 16..23;
        //          lane L>=16 -> M=L-16, halves 0..7 = K 8..15, 8..15 = K 24..31.
        // B 32x16: lane L -> N=L%16, halves in the same K order as A.
        const int row  = lane & 15;
        const int koff = (lane < 16) ? 0 : 8;
#pragma unroll
        for (int ku = 0; ku < 2; ++ku) {
            v8h lo = *(const v8h*)&Al[row * 64 + ku * 32 + koff];
            v8h hi = *(const v8h*)&Al[row * 64 + ku * 32 + 16 + koff];
            v16h a = __builtin_shufflevector(lo, hi, 0, 1, 2, 3, 4, 5, 6, 7,
                                             8, 9, 10, 11, 12, 13, 14, 15);
            const _Float16* brow = &Bl[wid][row * 64 + ku * 32 + koff];
            v8h b0l = *(const v8h*)&brow[0];
            v8h b0h = *(const v8h*)&brow[16];
            v8h b1l = *(const v8h*)&brow[16 * 64];
            v8h b1h = *(const v8h*)&brow[16 * 64 + 16];
            v16h b0 = __builtin_shufflevector(b0l, b0h, 0, 1, 2, 3, 4, 5, 6, 7,
                                              8, 9, 10, 11, 12, 13, 14, 15);
            v16h b1 = __builtin_shufflevector(b1l, b1h, 0, 1, 2, 3, 4, 5, 6, 7,
                                              8, 9, 10, 11, 12, 13, 14, 15);
            c0 = __builtin_amdgcn_wmma_f32_16x16x32_f16(
                false, a, false, b0, (short)0, c0, false, false);
            c1 = __builtin_amdgcn_wmma_f32_16x16x32_f16(
                false, a, false, b1, (short)0, c1, false, false);
        }
    }

    // C layout: VGPR r, lane L -> M = r + 8*(L>=16), N = L%16
    const int cc = lane & 15;
    const int mo = (lane >> 4) * 8;
#pragma unroll
    for (int t = 0; t < 2; ++t) {
        int n = nwave + t * 16 + cc;
        if (n < Ntot) {
            int b = n / HW, s = n % HW;
            const v8f& c = t ? c1 : c0;
#pragma unroll
            for (int r = 0; r < 8; ++r) {
                int co = mbase + mo + r;
                if (co < Cout)
                    out[((size_t)b * Cout + co) * HW + s] = c[r] + bias[co];
            }
        }
    }
}

// ---------------------------------------------------------------------------
// GroupNorm (32 groups) + optional SiLU. One block per (batch, group).
// ---------------------------------------------------------------------------
__global__ __launch_bounds__(256) void gn_silu_kernel(
    const float* __restrict__ in, const float* __restrict__ gam,
    const float* __restrict__ bet, float* __restrict__ out,
    int C, int HW, int do_silu)
{
    int b   = blockIdx.x >> 5;
    int grp = blockIdx.x & 31;
    int cg  = C / 32;
    int n   = cg * HW;
    const float* base = in + ((size_t)b * C + grp * cg) * HW;
    float s = 0.f, ss = 0.f;
    for (int i = threadIdx.x; i < n; i += 256) {
        float v = base[i];
        s += v; ss += v * v;
    }
    __shared__ float sh[256], sh2[256];
    sh[threadIdx.x] = s; sh2[threadIdx.x] = ss;
    __syncthreads();
    for (int o = 128; o > 0; o >>= 1) {
        if (threadIdx.x < o) {
            sh[threadIdx.x]  += sh[threadIdx.x + o];
            sh2[threadIdx.x] += sh2[threadIdx.x + o];
        }
        __syncthreads();
    }
    float mean = sh[0] / n;
    float var  = sh2[0] / n - mean * mean;
    float rstd = rsqrtf(var + 1e-5f);
    float* ob = out + ((size_t)b * C + grp * cg) * HW;
    for (int i = threadIdx.x; i < n; i += 256) {
        int c = grp * cg + i / HW;
        float v = (base[i] - mean) * rstd * gam[c] + bet[c];
        if (do_silu) v = v / (1.f + __expf(-v));
        ob[i] = v;
    }
}

// ---------------------------------------------------------------------------
// Attention core: qkv [B,384,S] channel-major. dh=8, 16 heads. One thread per
// (b, h, query); online softmax over S keys.
// ---------------------------------------------------------------------------
__global__ __launch_bounds__(256) void attn_kernel(
    const float* __restrict__ qkv, float* __restrict__ out, int B, int S)
{
    int t = blockIdx.x * 256 + threadIdx.x;
    if (t >= B * NH * S) return;
    int sq = t % S;
    int h  = (t / S) % NH;
    int b  = t / (S * NH);

    const float* qp = qkv + ((size_t)b * 384 + h * DH) * S + sq;
    const float* kb = qkv + ((size_t)b * 384 + 128 + h * DH) * S;
    const float* vb = qkv + ((size_t)b * 384 + 256 + h * DH) * S;

    float q[DH];
#pragma unroll
    for (int d = 0; d < DH; ++d) q[d] = qp[(size_t)d * S];

    const float scale = 0.3535533906f; // 1/sqrt(8)
    float m = -1e30f, l = 0.f, acc[DH] = {};
    for (int sk = 0; sk < S; ++sk) {
        float s = 0.f;
#pragma unroll
        for (int d = 0; d < DH; ++d) s += q[d] * kb[(size_t)d * S + sk];
        s *= scale;
        float nm   = fmaxf(m, s);
        float corr = __expf(m - nm);
        float wv   = __expf(s - nm);
        l = l * corr + wv;
#pragma unroll
        for (int d = 0; d < DH; ++d)
            acc[d] = acc[d] * corr + wv * vb[(size_t)d * S + sk];
        m = nm;
    }
    float* op = out + ((size_t)b * 128 + h * DH) * S + sq;
    float inv = 1.f / l;
#pragma unroll
    for (int d = 0; d < DH; ++d) op[(size_t)d * S] = acc[d] * inv;
}

// ---------------------------------------------------------------------------
// ConvTranspose2d k=4 s=2 p=1, weights [Cin,Cout,4,4]. One thread per output.
// out[ho,wo] += in[hi,wi]*w[ci,co,ky,kx] with hi=(ho+1-ky)/2 when even.
// ---------------------------------------------------------------------------
__global__ __launch_bounds__(256) void convt_kernel(
    const float* __restrict__ in, const float* __restrict__ w,
    const float* __restrict__ bias, float* __restrict__ out,
    int B, int Cin, int Hin, int Win, int Cout)
{
    int Ho = Hin * 2, Wo = Win * 2;
    long total = (long)B * Cout * Ho * Wo;
    long t = (long)blockIdx.x * 256 + threadIdx.x;
    if (t >= total) return;
    int wo = t % Wo;
    int ho = (t / Wo) % Ho;
    int co = (t / ((long)Wo * Ho)) % Cout;
    int b  = (int)(t / ((long)Wo * Ho * Cout));
    float acc = bias[co];
    for (int ky = 0; ky < 4; ++ky) {
        int th = ho + 1 - ky;
        if (th < 0 || (th & 1)) continue;
        int hi = th >> 1;
        if (hi >= Hin) continue;
        for (int kx = 0; kx < 4; ++kx) {
            int tw = wo + 1 - kx;
            if (tw < 0 || (tw & 1)) continue;
            int wi = tw >> 1;
            if (wi >= Win) continue;
            const float* ip = in + ((size_t)b * Cin) * Hin * Win + (size_t)hi * Win + wi;
            const float* wp = w + (size_t)co * 16 + (size_t)ky * 4 + kx;
            for (int ci = 0; ci < Cin; ++ci)
                acc += ip[(size_t)ci * Hin * Win] * wp[(size_t)ci * Cout * 16];
        }
    }
    out[t] = acc;
}

// ---------------------------------------------------------------------------
// Nearest-codebook quantize (K=20, C=3). Writes q (workspace + d_out section)
// and atomically accumulates both losses (equal in forward pass).
// ---------------------------------------------------------------------------
__global__ __launch_bounds__(256) void quantize_kernel(
    const float* __restrict__ z, const float* __restrict__ emb,
    float* __restrict__ qbuf, float* __restrict__ qout,
    float* __restrict__ cb, float* __restrict__ cm, int B, int HW)
{
    int t = blockIdx.x * 256 + threadIdx.x;
    if (t >= B * HW) return;
    int s = t % HW, b = t / HW;
    float zv[3];
#pragma unroll
    for (int c = 0; c < 3; ++c) zv[c] = z[((size_t)b * 3 + c) * HW + s];
    int best = 0; float bd = 1e30f;
    for (int kk = 0; kk < 20; ++kk) {
        float d = 0.f;
#pragma unroll
        for (int c = 0; c < 3; ++c) {
            float df = zv[c] - emb[kk * 3 + c];
            d += df * df;
        }
        if (d < bd) { bd = d; best = kk; }
    }
    float sse = 0.f;
#pragma unroll
    for (int c = 0; c < 3; ++c) {
        float qv = emb[best * 3 + c];
        size_t o = ((size_t)b * 3 + c) * HW + s;
        qbuf[o] = qv; qout[o] = qv;
        float df = qv - zv[c];
        sse += df * df;
    }
    float invn = 1.f / (float)(B * HW * 3);
    atomicAdd(cb, sse * invn);
    atomicAdd(cm, sse * invn);
}

__global__ void add_kernel(float* __restrict__ dst, const float* __restrict__ src, int n)
{
    int t = blockIdx.x * 256 + threadIdx.x;
    if (t < n) dst[t] += src[t];
}

__global__ void zero_kernel(float* p, int n)
{
    int t = blockIdx.x * 256 + threadIdx.x;
    if (t < n) p[t] = 0.f;
}

// ---------------------------------------------------------------------------
// Host orchestration
// ---------------------------------------------------------------------------
extern "C" void kernel_launch(void* const* d_in, const int* in_sizes, int n_in,
                              void* d_out, int out_size, void* d_ws, size_t ws_size,
                              hipStream_t stream)
{
    (void)in_sizes; (void)n_in; (void)out_size; (void)ws_size;
    const int B = 2;
    auto P = [&](int i) { return (const float*)d_in[i]; };

    // workspace buffers (floats); max activation = 2*64*160*160 = 3,276,800
    float* ws = (float*)d_ws;
    const size_t SZ = 3276800;
    float* T0  = ws;
    float* T1  = ws + SZ;
    float* T2  = ws + 2 * SZ;
    float* T3  = ws + 3 * SZ;
    float* QKV = ws + 4 * SZ;            // 2*384*1600 = 1,228,800
    float* ATT = QKV + 1228800;          // 2*128*1600 =   409,600

    float* dout = (float*)d_out;
    float* q_out = dout + 153600;
    float* cb    = dout + 163200;
    float* cm    = dout + 163201;

    auto conv = [&](const float* in, int wi, int bi, float* out,
                    int Cin, int H, int W, int Cout, int k, int pad) {
        int Ntot = B * H * W;
        dim3 grid((Ntot + 127) / 128, (Cout + 15) / 16);
        conv_wmma_kernel<<<grid, 128, 0, stream>>>(in, P(wi), P(bi), out,
                                                   B, Cin, H, W, Cout, k, pad);
    };
    auto gn = [&](const float* in, int gi, int bi, float* out, int C, int HW, int silu) {
        gn_silu_kernel<<<B * 32, 256, 0, stream>>>(in, P(gi), P(bi), out, C, HW, silu);
    };
    auto addk = [&](float* dst, const float* src, int n) {
        add_kernel<<<(n + 255) / 256, 256, 0, stream>>>(dst, src, n);
    };
    // resblock leaves at base: conv1.b, conv1.w, conv2.b, conv2.w,
    //                          gn1.b, gn1.g, gn2.b, gn2.g, res.b, res.w
    auto resblock = [&](const float* x, float* y, int base,
                        int Ci, int Co, int H, int W) {
        gn(x, base + 5, base + 4, T1, Ci, H * W, 1);
        conv(T1, base + 1, base + 0, T2, Ci, H, W, Co, 3, 1);
        gn(T2, base + 7, base + 6, T1, Co, H * W, 1);
        conv(T1, base + 3, base + 2, y, Co, H, W, Co, 3, 1);
        conv(x, base + 9, base + 8, T2, Ci, H, W, Co, 1, 0);
        addk(y, T2, B * Co * H * W);
    };
    auto attnblock = [&](float* y, int gnb, int gng,
                         int inb, int inw, int outb, int outw, int S) {
        gn(y, gng, gnb, T1, 128, S, 0);
        conv(T1, inw, inb, QKV, 128, S, 1, 384, 1, 0);
        attn_kernel<<<(B * NH * S + 255) / 256, 256, 0, stream>>>(QKV, ATT, B, S);
        conv(ATT, outw, outb, T2, 128, S, 1, 128, 1, 0);
        addk(y, T2, B * 128 * S);
    };

    // ---------------- encoder ----------------
    conv(P(0), 57, 56, T0, 3, 40, 40, 32, 3, 1);           // enc_in
    resblock(T0, T3, 33, 32, 64, 40, 40);                  // down[0]
    resblock(T3, T0, 43, 64, 128, 40, 40);                 // down[1]
    resblock(T0, T3, 64, 128, 128, 40, 40);                // enc_mid.res[0]
    attnblock(T3, 62, 63, 58, 59, 60, 61, 1600);           // enc_mid.attn[0]
    resblock(T3, T0, 74, 128, 128, 40, 40);                // enc_mid.res[1]
    gn(T0, 55, 54, T1, 128, 1600, 1);                      // enc_gn + silu
    conv(T1, 85, 84, T3, 128, 40, 40, 3, 3, 1);            // enc_out
    conv(T3, 89, 88, T0, 3, 40, 40, 3, 1, 0);              // pre_q -> z

    // ---------------- quantize ----------------
    zero_kernel<<<1, 32, 0, stream>>>(cb, 2);
    quantize_kernel<<<(B * 1600 + 255) / 256, 256, 0, stream>>>(
        T0, P(53), T3, q_out, cb, cm, B, 1600);

    // ---------------- decoder ----------------
    conv(T3, 87, 86, T0, 3, 40, 40, 3, 1, 0);              // post_q
    conv(T0, 4, 3, T3, 3, 40, 40, 128, 3, 1);              // dec_in
    resblock(T3, T0, 11, 128, 128, 40, 40);                // dec_mid.res[0]
    attnblock(T0, 9, 10, 5, 6, 7, 8, 1600);                // dec_mid.attn[0]
    resblock(T0, T3, 21, 128, 128, 40, 40);                // dec_mid.res[1]

    // up[0]: convT 128->128, 40->80; resblock 128->64 @80
    {
        long n = (long)B * 128 * 80 * 80;
        convt_kernel<<<(int)((n + 255) / 256), 256, 0, stream>>>(
            T3, P(101), P(100), T0, B, 128, 40, 40, 128);
    }
    resblock(T0, T3, 90, 128, 64, 80, 80);

    // up[1]: convT 64->64, 80->160; resblock 64->32 @160
    {
        long n = (long)B * 64 * 160 * 160;
        convt_kernel<<<(int)((n + 255) / 256), 256, 0, stream>>>(
            T3, P(113), P(112), T0, B, 64, 80, 80, 64);
    }
    resblock(T0, T3, 102, 64, 32, 160, 160);

    gn(T3, 2, 1, T1, 32, 25600, 1);                        // dec_gn + silu
    conv(T1, 32, 31, dout, 32, 160, 160, 3, 3, 1);         // dec_out -> d_out
}
